// InitialEdgeEmbedding_87900800680238
// MI455X (gfx1250) — matrix operand order repaired and checked by reading, not compiled, not run on hardware
//
#include <hip/hip_runtime.h>
#include <cstdint>
#include <cstddef>

// ---- Problem constants (from reference) ----
#define E_EDGES   800000
#define N_NODES   50000
#define ATOM_F    128
#define NUM_RAD   16
#define K_TOT     272            // 2*ATOM_F + NUM_RAD
#define K_PAD     288            // padded to 9 chunks of 32 (bf16 WMMA K)
#define K_ROW     296            // LDS row stride in elements (rows stay 16B aligned)
#define OUT_F     128
#define N_TILES   8              // OUT_F / 16
#define K_CHUNKS  9              // K_PAD / 32
#define WAVES_PB  8
#define TPB       (WAVES_PB * 32)
#define EDGES_PB  (WAVES_PB * 16)

typedef __attribute__((ext_vector_type(16))) __bf16 v16bf;
typedef __attribute__((ext_vector_type(2)))  __bf16 v2bf;
typedef __attribute__((ext_vector_type(8)))  float  v8f;

#if defined(__has_builtin)
#if __has_builtin(__builtin_amdgcn_global_load_async_to_lds_b128)
#define HAVE_ASYNC_LDS 1
#endif
#if __has_builtin(__builtin_amdgcn_cvt_pk_bf16_f32)
#define HAVE_CVT_PK_BF16 1
#endif
#endif

#ifdef HAVE_ASYNC_LDS
// Exact parameter element type per hipcc diagnostic:
// "__attribute__((__vector_size__(4 * sizeof(int)))) int __device__ *"
typedef int v4i_ __attribute__((vector_size(4 * sizeof(int))));
typedef __attribute__((address_space(1))) v4i_* gl_v4i_p;  // global (__device__)
typedef __attribute__((address_space(3))) v4i_* sh_v4i_p;  // LDS (__shared__)
#endif

// float -> bf16 bits, round-to-nearest-even (scalar fallback)
__device__ __forceinline__ unsigned int f2bf(float f) {
  unsigned int u = __float_as_uint(f);
  return (u + 0x7FFFu + ((u >> 16) & 1u)) >> 16;
}
// two f32 -> packed bf16x2 (single v_cvt_pk_bf16_f32 when available)
__device__ __forceinline__ unsigned int pkbf(float lo, float hi) {
#ifdef HAVE_CVT_PK_BF16
  union { v2bf v; unsigned int u; } c;
  c.v = __builtin_amdgcn_cvt_pk_bf16_f32(lo, hi);
  return c.u;
#else
  return f2bf(lo) | (f2bf(hi) << 16);
#endif
}

// two 16B LDS loads -> one 16-element bf16 WMMA fragment
__device__ __forceinline__ v16bf load_frag(const unsigned short* p0,
                                           const unsigned short* p1) {
  union { uint4 q[2]; v16bf v; } u;
  u.q[0] = *(const uint4*)p0;
  u.q[1] = *(const uint4*)p1;
  return u.v;
}

// W [K_TOT][OUT_F] f32 (row-major)  ->  Wt [OUT_F][K_ROW] bf16 (transposed, K zero-padded)
__global__ __launch_bounds__(256) void prep_w_kernel(const float* __restrict__ W,
                                                     unsigned short* __restrict__ Wt) {
  int i = blockIdx.x * blockDim.x + threadIdx.x;
  if (i >= OUT_F * K_ROW) return;
  int n = i / K_ROW;
  int k = i - n * K_ROW;
  unsigned short v = 0;
  if (k < K_TOT) v = (unsigned short)f2bf(W[(size_t)k * OUT_F + n]);
  Wt[i] = v;
}

__global__ __launch_bounds__(TPB) void edge_embed_kernel(
    const float* __restrict__ x, const float* __restrict__ pos,
    const int* __restrict__ ei, const unsigned short* __restrict__ Wt,
    float* __restrict__ out) {
  extern __shared__ __align__(16) unsigned char smem[];
  unsigned short* Wl = (unsigned short*)smem;          // [OUT_F][K_ROW] bf16
  unsigned short* Al = Wl + OUT_F * K_ROW;             // 8 x [16][K_ROW] bf16

  const int tid  = threadIdx.x;
  const int wave = tid >> 5;
  const int lane = tid & 31;
  unsigned short* Aw = Al + wave * (16 * K_ROW);

  // ---- Stage bf16 W (transposed, padded) into LDS ----
  {
    const uint4* wsrc = (const uint4*)Wt;
    uint4* wdst = (uint4*)Wl;
    const int nch = (OUT_F * K_ROW * 2) / 16;  // 4736 16-byte chunks
#ifdef HAVE_ASYNC_LDS
    for (int i = tid; i < nch; i += TPB) {
      __builtin_amdgcn_global_load_async_to_lds_b128(
          (gl_v4i_p)(wsrc + i), (sh_v4i_p)(wdst + i), 0, 0);
    }
#if __has_builtin(__builtin_amdgcn_s_wait_asynccnt)
    __builtin_amdgcn_s_wait_asynccnt(0);
#else
    asm volatile("s_wait_asynccnt 0" ::: "memory");
#endif
#else
    for (int i = tid; i < nch; i += TPB) wdst[i] = wsrc[i];
#endif
  }

  // ---- This wave's 16-edge tile ----
  const int tile  = blockIdx.x * WAVES_PB + wave;  // grid is exact: E/16 tiles
  const int ebase = tile * 16;

  int srcIdx = 0, dstIdx = 0;
  if (lane < 16) {
    srcIdx = ei[ebase + lane];
    dstIdx = ei[E_EDGES + ebase + lane];
  }

  // ---- Bessel RBF + K padding: 2 lanes per edge (8 radials each) ----
  {
    const int es = lane >> 1, half = lane & 1;
    int s = __shfl(srcIdx, es);
    int t = __shfl(dstIdx, es);
    float dx = pos[3 * s + 0] - pos[3 * t + 0];
    float dy = pos[3 * s + 1] - pos[3 * t + 1];
    float dz = pos[3 * s + 2] - pos[3 * t + 2];
    float r = sqrtf(dx * dx + dy * dy + dz * dz);
    float d = fmaxf(r * (1.0f / 6.0f), 1e-6f);
    float env = 0.0f;
    if (d < 1.0f) {  // envelope p=5: 1/d - 21 d^4 + 35 d^5 - 15 d^6
      float d2 = d * d, d4 = d2 * d2;
      env = 1.0f / d + d4 * (-21.0f + d * (35.0f - 15.0f * d));
    }
    float vals[8];
#pragma unroll
    for (int j = 0; j < 8; ++j) {
      float f = (float)(half * 8 + j + 1) * 3.14159265358979323f;
      vals[j] = env * __sinf(f * d);
    }
    uint4 q;
    q.x = pkbf(vals[0], vals[1]);
    q.y = pkbf(vals[2], vals[3]);
    q.z = pkbf(vals[4], vals[5]);
    q.w = pkbf(vals[6], vals[7]);
    *(uint4*)(Aw + es * K_ROW + 2 * ATOM_F + half * 8) = q;                 // rbf
    *(uint4*)(Aw + es * K_ROW + K_TOT + half * 8) = make_uint4(0, 0, 0, 0); // K pad
  }

  // ---- Gather x[src], x[dst] rows -> bf16 A tile (whole wave per row) ----
#pragma unroll 4
  for (int r = 0; r < 16; ++r) {
    int s = __shfl(srcIdx, r);
    int t = __shfl(dstIdx, r);
    const float2* xs = (const float2*)(x + (size_t)s * ATOM_F);
    const float2* xt = (const float2*)(x + (size_t)t * ATOM_F);
    float2 a0 = xs[lane], a1 = xs[lane + 32];
    float2 b0 = xt[lane], b1 = xt[lane + 32];
    unsigned int* dst = (unsigned int*)(Aw + r * K_ROW);
    dst[lane]      = pkbf(a0.x, a0.y);  // h_a cols 0..63
    dst[lane + 32] = pkbf(a1.x, a1.y);  // h_a cols 64..127
    dst[64 + lane] = pkbf(b0.x, b0.y);  // h_c cols 128..191
    dst[96 + lane] = pkbf(b1.x, b1.y);  // h_c cols 192..255
  }

  __syncthreads();

  // ---- [16 x 288]bf16 x [288 x 128]bf16 -> [16 x 128]f32 via v_wmma ----
  v8f acc[N_TILES];
  const v8f vzero = {0.f, 0.f, 0.f, 0.f, 0.f, 0.f, 0.f, 0.f};
#pragma unroll
  for (int n = 0; n < N_TILES; ++n) acc[n] = vzero;

  const int ar   = lane & 15;
  const int hi8  = (lane & 16) >> 1;  // 0 or 8
  const int hi16 = (lane & 16);       // 0 or 16

  for (int kc = 0; kc < K_CHUNKS; ++kc) {
    // A frag: lanes 0-15 -> K {0..7,16..23}, lanes 16-31 -> K {8..15,24..31}
    const unsigned short* ap = Aw + ar * K_ROW + kc * 32 + hi8;
    v16bf afrag = load_frag(ap, ap + 16);
#pragma unroll
    for (int n = 0; n < N_TILES; ++n) {
      // B frag: lane = column N, lanes 0-15 -> K {0..7,16..23}, lanes 16-31 -> K {8..15,24..31}
      const unsigned short* bp = Wl + (n * 16 + ar) * K_ROW + kc * 32 + hi8;
      v16bf bfrag = load_frag(bp, bp + 16);
      acc[n] = __builtin_amdgcn_wmma_f32_16x16x32_bf16(
          false, afrag, false, bfrag, (short)0, acc[n], false, false);
    }
  }

  // ---- SiLU epilogue + store (C layout: VGPR v -> M = v + hi8, lane -> N) ----
#pragma unroll
  for (int n = 0; n < N_TILES; ++n) {
    float* orow = out + (size_t)(ebase + hi8) * OUT_F + n * 16 + ar;
#pragma unroll
    for (int v = 0; v < 8; ++v) {
      float val = acc[n][v];
      float sg = val * __builtin_amdgcn_rcpf(1.0f + __expf(-val));
      orow[(size_t)v * OUT_F] = sg;
    }
  }
}

extern "C" void kernel_launch(void* const* d_in, const int* in_sizes, int n_in,
                              void* d_out, int out_size, void* d_ws, size_t ws_size,
                              hipStream_t stream) {
  (void)in_sizes; (void)n_in; (void)out_size; (void)ws_size;
  const float* x   = (const float*)d_in[0];
  const float* pos = (const float*)d_in[1];
  const int*   ei  = (const int*)d_in[2];
  const float* W   = (const float*)d_in[3];
  unsigned short* Wt = (unsigned short*)d_ws;  // 128*296*2 = 75776 bytes
  float* out = (float*)d_out;

  const int prepN = OUT_F * K_ROW;
  prep_w_kernel<<<(prepN + 255) / 256, 256, 0, stream>>>(W, Wt);

  const size_t lds_bytes =
      (size_t)(OUT_F * K_ROW + WAVES_PB * 16 * K_ROW) * sizeof(unsigned short);
  edge_embed_kernel<<<E_EDGES / EDGES_PB, TPB, lds_bytes, stream>>>(x, pos, ei, Wt, out);
}
